// Attention_85633057947969
// MI455X (gfx1250) — 8-bit matrix operand order repaired and checked by reading, compiled, not run on hardware
//
#include <hip/hip_runtime.h>
#include <hip/hip_bf16.h>
#include <math.h>

typedef __attribute__((ext_vector_type(16))) int   v16i;
typedef __attribute__((ext_vector_type(4)))  int   v4i;
typedef __attribute__((ext_vector_type(8)))  float v8f;

#define B_   16
#define S_   4096
#define H_   1024
#define CT   128
#define RT   32
#define NT   256
#define WSP  1040
#define SCH  4
#define SCL  (S_ / SCH)
#define ITERS (SCL / RT)

#if __has_builtin(__builtin_amdgcn_cvt_pk_fp8_f32)
static __device__ __forceinline__ unsigned pack4_fp8(float a, float b, float c, float d) {
  int v = __builtin_amdgcn_cvt_pk_fp8_f32(a, b, 0, false);
  v     = __builtin_amdgcn_cvt_pk_fp8_f32(c, d, v, true);
  return (unsigned)v;
}
#else
static __device__ __forceinline__ unsigned f2fp8(float x) {
  unsigned u = __float_as_uint(x);
  unsigned s = (u >> 24) & 0x80u;
  float a = fabsf(x);
  if (a >= 448.0f) return s | 0x7Eu;
  unsigned bits = __float_as_uint(a) + (1u << 19);
  int e = (int)(bits >> 23) - 127 + 7;
  if (e <= 0) return s;
  return s | ((unsigned)e << 3) | ((bits >> 20) & 7u);
}
static __device__ __forceinline__ unsigned pack4_fp8(float a, float b, float c, float d) {
  return f2fp8(a) | (f2fp8(b) << 8) | (f2fp8(c) << 16) | (f2fp8(d) << 24);
}
#endif

static __device__ __forceinline__ float fast_tanh(float x) {
#if __has_builtin(__builtin_amdgcn_tanhf)
  return __builtin_amdgcn_tanhf(x);
#else
  float y;
  asm("v_tanh_f32 %0, %1" : "=v"(y) : "v"(x));
  return y;
#endif
}

__global__ __launch_bounds__(256)
void wconv_kernel(const float* __restrict__ W, unsigned char* __restrict__ Wf8) {
  int i = blockIdx.x * blockDim.x + threadIdx.x;
  const float* p = W + (size_t)i * 8;
  float4 f0 = *(const float4*)p;
  float4 f1 = *(const float4*)(p + 4);
  uint2 d;
  d.x = pack4_fp8(f0.x, f0.y, f0.z, f0.w);
  d.y = pack4_fp8(f1.x, f1.y, f1.z, f1.w);
  *(uint2*)(Wf8 + (size_t)i * 8) = d;
}

__global__ __launch_bounds__(NT)
void attn_kernel(const float* __restrict__ x, const unsigned char* __restrict__ Wf8,
                 const float* __restrict__ bias,
                 float* __restrict__ Pm, float* __restrict__ Pz, float* __restrict__ Pn) {
  __shared__ __align__(16) unsigned char Ws[CT * WSP];
  __shared__ __align__(16) unsigned char As[2][RT * WSP];
  __shared__ float MrgM[2][8][16];
  __shared__ float MrgZ[2][8][16];
  __shared__ float MrgN[2][8][16];

  const int tid  = threadIdx.x;
  const int lane = tid & 31;
  const int wave = tid >> 5;
  const int wm   = wave & 1;
  const int wg   = wave >> 1;
  const int half = lane >> 4;
  const int l15  = lane & 15;

  const int ctile   = blockIdx.x & 7;
  const int b       = (blockIdx.x >> 3) & 15;
  const int sc      = blockIdx.x >> 7;
  const int ch_base = ctile * CT;

  for (int i = 0; i < 32; ++i) {
    int j   = i * NT + tid;
    int row = j >> 6;
    int c16 = j & 63;
    const unsigned char* g = Wf8 + (size_t)(ch_base + row) * H_ + c16 * 16;
    unsigned l = (unsigned)(size_t)&Ws[row * WSP + c16 * 16];
    asm volatile("global_load_async_to_lds_b128 %0, %1, off"
                 :: "v"(l), "v"(g) : "memory");
  }

  const int   chg0 = ch_base + wg * 32 + l15;
  const int   chg1 = chg0 + 16;
  const float eb0  = bias[chg0];
  const float eb1  = bias[chg1];

  float m0 = -INFINITY, Z0 = 0.0f, N0 = 0.0f;
  float m1 = -INFINITY, Z1 = 0.0f, N1 = 0.0f;

  const float* xbase = x + ((size_t)b * S_ + (size_t)sc * SCL) * H_;

  {
    const float* xb = xbase;
    #pragma unroll
    for (int i = 0; i < (RT * H_ / 8) / NT; ++i) {
      int c   = i * NT + tid;
      int row = c >> 7;
      int kc  = c & 127;
      int k   = kc << 3;
      unsigned off = row * WSP + (kc >> 4) * 128 + ((kc >> 1) & 1) * 64
                   + ((kc >> 2) & 3) * 16 + (kc & 1) * 8;
      const float* p = xb + row * H_ + k;
      float4 f0 = *(const float4*)p;
      float4 f1 = *(const float4*)(p + 4);
      uint2 d;
      d.x = pack4_fp8(f0.x, f0.y, f0.z, f0.w);
      d.y = pack4_fp8(f1.x, f1.y, f1.z, f1.w);
      *(uint2*)(As[0] + off) = d;
    }
  }
  asm volatile("s_wait_asynccnt 0" ::: "memory");
  __syncthreads();

  const unsigned aroff  = (unsigned)(wm * 16 + l15) * WSP + (unsigned)half * 64;
  const unsigned broff0 = (unsigned)(wg * 32 + l15) * WSP + (unsigned)half * 16;
  const unsigned broff1 = (unsigned)(wg * 32 + 16 + l15) * WSP + (unsigned)half * 16;

  for (int it = 0; it < ITERS; ++it) {
    const int cur = it & 1;

    v8f acc0 = {};
    v8f acc1 = {};
    const unsigned char* Ab = As[cur];
    #pragma unroll
    for (int kb = 0; kb < 8; ++kb) {
      union { v16i v; v4i q[4]; } A, B0, B1;
      const unsigned char* ap = Ab + aroff + kb * 128;
      A.q[0] = *(const v4i*)(ap);
      A.q[1] = *(const v4i*)(ap + 16);
      A.q[2] = *(const v4i*)(ap + 32);
      A.q[3] = *(const v4i*)(ap + 48);
      const unsigned char* bp0 = Ws + broff0 + kb * 128;
      B0.q[0] = *(const v4i*)(bp0);
      B0.q[1] = *(const v4i*)(bp0 + 32);
      B0.q[2] = *(const v4i*)(bp0 + 64);
      B0.q[3] = *(const v4i*)(bp0 + 96);
      const unsigned char* bp1 = Ws + broff1 + kb * 128;
      B1.q[0] = *(const v4i*)(bp1);
      B1.q[1] = *(const v4i*)(bp1 + 32);
      B1.q[2] = *(const v4i*)(bp1 + 64);
      B1.q[3] = *(const v4i*)(bp1 + 96);
      acc0 = __builtin_amdgcn_wmma_f32_16x16x128_fp8_fp8(
                 A.v, B0.v, (short)0, acc0, false, false);
      acc1 = __builtin_amdgcn_wmma_f32_16x16x128_fp8_fp8(
                 A.v, B1.v, (short)0, acc1, false, false);
    }

    const int    rbase = wm * 16 + half * 8;
    const float* xrow  = xbase + ((size_t)it * RT + rbase) * H_;
    {
      float er[8], xr[8];
      float tmax = -INFINITY;
      #pragma unroll
      for (int r = 0; r < 8; ++r) {
        float e = fast_tanh(acc0[r] + eb0);
        er[r] = e;
        tmax  = fmaxf(tmax, e);
        xr[r] = xrow[(size_t)r * H_ + chg0];
      }
      float mn = fmaxf(m0, tmax);
      float sc2 = __expf(m0 - mn);
      Z0 *= sc2; N0 *= sc2;
      #pragma unroll
      for (int r = 0; r < 8; ++r) {
        float p = __expf(er[r] - mn);
        Z0 += p;
        N0 += p * xr[r];
      }
      m0 = mn;
    }
    {
      float er[8], xr[8];
      float tmax = -INFINITY;
      #pragma unroll
      for (int r = 0; r < 8; ++r) {
        float e = fast_tanh(acc1[r] + eb1);
        er[r] = e;
        tmax  = fmaxf(tmax, e);
        xr[r] = xrow[(size_t)r * H_ + chg1];
      }
      float mn = fmaxf(m1, tmax);
      float sc2 = __expf(m1 - mn);
      Z1 *= sc2; N1 *= sc2;
      #pragma unroll
      for (int r = 0; r < 8; ++r) {
        float p = __expf(er[r] - mn);
        Z1 += p;
        N1 += p * xr[r];
      }
      m1 = mn;
    }

    if (it + 1 < ITERS) {
      const float* xb = xbase + (size_t)(it + 1) * RT * H_;
      unsigned char* dst = As[1 - cur];
      #pragma unroll
      for (int i = 0; i < (RT * H_ / 8) / NT; ++i) {
        int c   = i * NT + tid;
        int row = c >> 7;
        int kc  = c & 127;
        int k   = kc << 3;
        unsigned off = row * WSP + (kc >> 4) * 128 + ((kc >> 1) & 1) * 64
                     + ((kc >> 2) & 3) * 16 + (kc & 1) * 8;
        const float* p = xb + row * H_ + k;
        float4 f0 = *(const float4*)p;
        float4 f1 = *(const float4*)(p + 4);
        uint2 d;
        d.x = pack4_fp8(f0.x, f0.y, f0.z, f0.w);
        d.y = pack4_fp8(f1.x, f1.y, f1.z, f1.w);
        *(uint2*)(dst + off) = d;
      }
    }
    __syncthreads();
  }

  {
    float ma = __shfl_xor(m0, 16, 32);
    float Za = __shfl_xor(Z0, 16, 32);
    float Na = __shfl_xor(N0, 16, 32);
    float M  = fmaxf(m0, ma);
    float sA = __expf(m0 - M), sB = __expf(ma - M);
    if (lane < 16) {
      MrgM[wm][wg * 2 + 0][l15] = M;
      MrgZ[wm][wg * 2 + 0][l15] = Z0 * sA + Za * sB;
      MrgN[wm][wg * 2 + 0][l15] = N0 * sA + Na * sB;
    }
  }
  {
    float ma = __shfl_xor(m1, 16, 32);
    float Za = __shfl_xor(Z1, 16, 32);
    float Na = __shfl_xor(N1, 16, 32);
    float M  = fmaxf(m1, ma);
    float sA = __expf(m1 - M), sB = __expf(ma - M);
    if (lane < 16) {
      MrgM[wm][wg * 2 + 1][l15] = M;
      MrgZ[wm][wg * 2 + 1][l15] = Z1 * sA + Za * sB;
      MrgN[wm][wg * 2 + 1][l15] = N1 * sA + Na * sB;
    }
  }
  __syncthreads();

  if (wm == 0 && lane < 16) {
    #pragma unroll
    for (int g = 0; g < 2; ++g) {
      int   gi = wg * 2 + g;
      float Ma = MrgM[0][gi][l15], Mb = MrgM[1][gi][l15];
      float M  = fmaxf(Ma, Mb);
      float sa = __expf(Ma - M), sb = __expf(Mb - M);
      float Z  = MrgZ[0][gi][l15] * sa + MrgZ[1][gi][l15] * sb;
      float N  = MrgN[0][gi][l15] * sa + MrgN[1][gi][l15] * sb;
      int ch = ch_base + gi * 16 + l15;
      size_t p = ((size_t)sc * B_ + b) * H_ + ch;
      Pm[p] = M; Pz[p] = Z; Pn[p] = N;
    }
  }
}

__global__ __launch_bounds__(256)
void combine_kernel(const float* __restrict__ Pm, const float* __restrict__ Pz,
                    const float* __restrict__ Pn, float* __restrict__ out) {
  int idx = blockIdx.x * blockDim.x + threadIdx.x;
  float M = -INFINITY;
  #pragma unroll
  for (int s = 0; s < SCH; ++s) M = fmaxf(M, Pm[(size_t)s * B_ * H_ + idx]);
  float Z = 0.0f, N = 0.0f;
  #pragma unroll
  for (int s = 0; s < SCH; ++s) {
    float w = __expf(Pm[(size_t)s * B_ * H_ + idx] - M);
    Z += Pz[(size_t)s * B_ * H_ + idx] * w;
    N += Pn[(size_t)s * B_ * H_ + idx] * w;
  }
  out[idx] = N / Z;
}

extern "C" void kernel_launch(void* const* d_in, const int* in_sizes, int n_in,
                              void* d_out, int out_size, void* d_ws, size_t ws_size,
                              hipStream_t stream) {
  const float* x    = (const float*)d_in[0];
  const float* W    = (const float*)d_in[1];
  const float* bias = (const float*)d_in[2];
  float*       out  = (float*)d_out;

  unsigned char* Wf8 = (unsigned char*)d_ws;
  float* Pm = (float*)((char*)d_ws + (1u << 20));
  float* Pz = Pm + (size_t)SCH * B_ * H_;
  float* Pn = Pz + (size_t)SCH * B_ * H_;

  wconv_kernel<<<(H_ * H_ / 8) / 256, 256, 0, stream>>>(W, Wf8);
  attn_kernel<<<(H_ / CT) * B_ * SCH, NT, 0, stream>>>(x, Wf8, bias, Pm, Pz, Pn);
  combine_kernel<<<(B_ * H_) / 256, 256, 0, stream>>>(Pm, Pz, Pn, out);
}
